// MSABlock_22351009808995
// MI455X (gfx1250) — compile-verified, matrix operand order checked by reading, not run on hardware
//
#include <hip/hip_runtime.h>

typedef unsigned short u16;
typedef __attribute__((ext_vector_type(16))) __bf16 v16bf;
typedef __attribute__((ext_vector_type(8)))  float  v8f;

#define BATCH 32
#define HH 56
#define WW 56
#define CH 192
#define HEADS 6
#define HD 32
#define WSZ 7
#define SHIFT 3
#define NTOK 49
#define NWIN 2048          // 32 * 8 * 8
#define ROWS 100352        // NWIN * 49
#define QSCALE 0.17677669529663687f

union Frag { v16bf v; unsigned u[8]; };

__device__ inline u16 f2bf(float f) {
  unsigned u = __float_as_uint(f);
  unsigned r = u + 0x7FFFu + ((u >> 16) & 1u);
  return (u16)(r >> 16);
}

// gfx1250 async global->LDS copy (ASYNCcnt-tracked; per-lane LDS dest addr)
__device__ inline void async_b128(unsigned lds_off, const void* gaddr) {
  asm volatile("global_load_async_to_lds_b128 %0, %1, off"
               :: "v"(lds_off), "v"(gaddr) : "memory");
}
__device__ inline void wait_async0() {
  asm volatile("s_wait_asynccnt 0x0" ::: "memory");
}
__device__ inline unsigned lds_addr(const void* p) {
  return (unsigned)(size_t)p;   // flat LDS pointer: low 32 bits = LDS offset
}

// Load a 16x32 bf16 fragment (A-operand layout; B uses same K-pair layout on
// an N-major tile). base points at tile origin; stride = row pitch in u16.
__device__ inline v16bf load_frag(const u16* base, int stride, int lane) {
  Frag f;
  int mn = lane & 15;
  int kh = (lane & 16) ? 8 : 0;
  const u16* row = base + mn * stride;
#pragma unroll
  for (int j = 0; j < 4; ++j) {
    f.u[j]     = *(const unsigned*)(row + kh + 2 * j);        // K = kh..kh+7
    f.u[4 + j] = *(const unsigned*)(row + 16 + kh + 2 * j);   // K = 16+kh..
  }
  return f.v;
}

__device__ inline v8f wmma_bf16(v16bf a, v16bf b, v8f c) {
  return __builtin_amdgcn_wmma_f32_16x16x32_bf16(false, a, false, b, (short)0, c,
                                                 false, false);
}

// ---------------------------------------------------------------------------
// K1: LayerNorm + roll(-3,-3); writes fp32 shifted image + bf16 window layout
// ---------------------------------------------------------------------------
__global__ __launch_bounds__(256) void ln_shift_kernel(
    const float* __restrict__ x, const float* __restrict__ gamma,
    const float* __restrict__ beta, float* __restrict__ xs,
    u16* __restrict__ xw) {
  int wv = threadIdx.x >> 5, lane = threadIdx.x & 31;
  int p = blockIdx.x * 8 + wv;                     // pixel id, grid exact
  const float* px = x + (size_t)p * CH;
  float v[6];
  float s = 0.f;
#pragma unroll
  for (int i = 0; i < 6; ++i) { v[i] = px[lane + 32 * i]; s += v[i]; }
#pragma unroll
  for (int m = 16; m >= 1; m >>= 1) s += __shfl_xor(s, m, 32);
  float mu = s * (1.f / CH);
  float s2 = 0.f;
#pragma unroll
  for (int i = 0; i < 6; ++i) { float d = v[i] - mu; s2 += d * d; }
#pragma unroll
  for (int m = 16; m >= 1; m >>= 1) s2 += __shfl_xor(s2, m, 32);
  float inv = rsqrtf(s2 * (1.f / CH) + 1e-5f);

  int b = p / (HH * WW), rem = p % (HH * WW), h = rem / WW, w = rem % WW;
  int hs = h - SHIFT; if (hs < 0) hs += HH;
  int wsC = w - SHIFT; if (wsC < 0) wsC += WW;
  size_t so = ((size_t)(b * HH + hs) * WW + wsC) * CH;
  int win = (b * 8 + hs / WSZ) * 8 + wsC / WSZ;
  int tok = (hs % WSZ) * WSZ + (wsC % WSZ);
  u16* pw = xw + ((size_t)win * NTOK + tok) * CH;
#pragma unroll
  for (int i = 0; i < 6; ++i) {
    int c = lane + 32 * i;
    float y = (v[i] - mu) * inv * gamma[c] + beta[c];
    xs[so + c] = y;
    pw[c] = f2bf(y);
  }
}

// ---------------------------------------------------------------------------
// K2: fp32 -> bf16 weight conversion
// ---------------------------------------------------------------------------
__global__ __launch_bounds__(256) void convert_w_kernel(
    const float* __restrict__ qw, const float* __restrict__ pw,
    u16* __restrict__ qwb, u16* __restrict__ pwb) {
  int i = blockIdx.x * 256 + threadIdx.x;
  if (i < 3 * CH * CH) qwb[i] = f2bf(qw[i]);
  if (i < CH * CH)     pwb[i] = f2bf(pw[i]);
}

// ---------------------------------------------------------------------------
// K3: depthwise 7x7 SAME conv on shifted image; output in window layout
// ---------------------------------------------------------------------------
__global__ __launch_bounds__(256) void dwconv_kernel(
    const float* __restrict__ xs, const float* __restrict__ dw_w,
    const float* __restrict__ dw_b, float* __restrict__ dww) {
  __shared__ __align__(16) float lH[13 * 13 * 16];
  __shared__ __align__(16) float lW[49 * 16];
  __shared__ float lBias[16];
  int win = blockIdx.x;
  int b = win >> 6, wi = win & 63, wr = wi >> 3, wc = wi & 7;
  int hs0 = wr * WSZ, ws0 = wc * WSZ;
  for (int c0 = 0; c0 < CH; c0 += 16) {
    for (int idx = threadIdx.x; idx < 13 * 13 * 16; idx += 256) {
      int cell = idx >> 4, cc = idx & 15;
      int hh = hs0 - 3 + cell / 13, ww2 = ws0 - 3 + cell % 13;
      float val = 0.f;
      if (hh >= 0 && hh < HH && ww2 >= 0 && ww2 < WW)
        val = xs[(((size_t)b * HH + hh) * WW + ww2) * CH + c0 + cc];
      lH[idx] = val;
    }
    for (int idx = threadIdx.x; idx < 49 * 16; idx += 256)
      lW[idx] = dw_w[(idx >> 4) * CH + c0 + (idx & 15)];
    if (threadIdx.x < 16) lBias[threadIdx.x] = dw_b[c0 + threadIdx.x];
    __syncthreads();
    for (int idx = threadIdx.x; idx < NTOK * 16; idx += 256) {
      int tok = idx >> 4, cc = idx & 15;
      int py = tok / WSZ, pxl = tok % WSZ;
      float acc = lBias[cc];
#pragma unroll
      for (int ky = 0; ky < 7; ++ky)
#pragma unroll
        for (int kx = 0; kx < 7; ++kx)
          acc += lH[((py + ky) * 13 + pxl + kx) * 16 + cc] *
                 lW[(ky * 7 + kx) * 16 + cc];
      dww[((size_t)win * NTOK + tok) * CH + c0 + cc] = acc;
    }
    __syncthreads();
  }
}

// ---------------------------------------------------------------------------
// K4: QKV GEMM [ROWS,192] x [192,576] -> scattered q/k/v (bf16), bias+scale
//     Double-buffered LDS tiles filled with async global->LDS copies.
// ---------------------------------------------------------------------------
__global__ __launch_bounds__(256) void qkv_gemm_kernel(
    const u16* __restrict__ xw, const u16* __restrict__ wq,
    const float* __restrict__ qkv_b, u16* __restrict__ qbuf,
    u16* __restrict__ kbuf, u16* __restrict__ vbuf) {
  __shared__ __align__(16) u16 lA[2][128 * 32];
  __shared__ __align__(16) u16 lB[2][64 * 32];
  int mBase = blockIdx.x * 128;
  int nBase = blockIdx.y * 64;
  int tid = threadIdx.x, lane = tid & 31, wv = tid >> 5;
  v8f acc[4];
#pragma unroll
  for (int t = 0; t < 4; ++t) acc[t] = v8f{0.f, 0.f, 0.f, 0.f, 0.f, 0.f, 0.f, 0.f};

  auto stage = [&](int k0, u16* dA, u16* dB) {
    // A tile: 128x32 bf16 = 512 16B chunks, 2 per thread
#pragma unroll
    for (int i = 0; i < 2; ++i) {
      int c = tid * 2 + i;
      int row = c >> 2, seg = (c & 3) * 8;
      const u16* src = xw + (size_t)(mBase + row) * CH + k0 + seg;
      async_b128(lds_addr(dA + row * 32 + seg), src);
      if (k0 + 64 < CH) __builtin_prefetch(src + 64, 0, 1);
    }
    // B tile: 64x32 bf16 = 256 16B chunks, 1 per thread
    int rB = tid >> 2, segB = (tid & 3) * 8;
    async_b128(lds_addr(dB + rB * 32 + segB),
               wq + (size_t)(nBase + rB) * CH + k0 + segB);
  };

  stage(0, lA[0], lB[0]);
  wait_async0();
  __syncthreads();
#pragma unroll
  for (int ks = 0; ks < 6; ++ks) {
    int buf = ks & 1;
    if (ks < 5) stage((ks + 1) * 32, lA[buf ^ 1], lB[buf ^ 1]);
    v16bf a = load_frag(lA[buf] + wv * 16 * 32, 32, lane);
#pragma unroll
    for (int t = 0; t < 4; ++t) {
      v16bf bfr = load_frag(lB[buf] + t * 16 * 32, 32, lane);
      acc[t] = wmma_bf16(a, bfr, acc[t]);
    }
    if (ks < 5) { wait_async0(); __syncthreads(); }
  }

  int m0 = mBase + wv * 16;
  int mh = (lane >> 4) << 3;
  int nl = lane & 15;
#pragma unroll
  for (int t = 0; t < 4; ++t) {
    int nG = nBase + t * 16 + nl;
    int qkvi = nG / CH, rem = nG % CH, head = rem >> 5, d = rem & 31;
    float bias = qkv_b[nG];
    u16* dst = (qkvi == 0) ? qbuf : ((qkvi == 1) ? kbuf : vbuf);
#pragma unroll
    for (int r = 0; r < 8; ++r) {
      int row = m0 + mh + r;
      int win = row / NTOK, tok = row % NTOK;
      float val = acc[t][r] + bias;
      if (qkvi == 0) val *= QSCALE;
      dst[(((size_t)win * HEADS + head) * NTOK + tok) * HD + d] = f2bf(val);
    }
  }
}

// ---------------------------------------------------------------------------
// K5: attention per (window, head): softmax(q k^T + rpb + mask) v + dw branch
// ---------------------------------------------------------------------------
__global__ __launch_bounds__(128) void attention_kernel(
    const u16* __restrict__ qb, const u16* __restrict__ kb,
    const u16* __restrict__ vb, const float* __restrict__ rpb,
    const float* __restrict__ mask, const float* __restrict__ dww,
    u16* __restrict__ pin) {
  __shared__ __align__(16) u16 lQ[64 * 32];
  __shared__ __align__(16) u16 lK[64 * 32];
  __shared__ __align__(16) u16 lVT[32 * 64];
  __shared__ __align__(16) u16 lP[4 * 16 * 64];

  int unit = blockIdx.x;
  int win = unit / HEADS, head = unit % HEADS;
  size_t base = ((size_t)win * HEADS + head) * NTOK * HD;
  int tid = threadIdx.x, lane = tid & 31, wv = tid >> 5;

  // async-stage q and k (49 rows x 64B = 196 16B chunks each).
  // Rows 49..63 stay garbage: every S element with m>=49 or n>=49 is
  // overwritten with -1e30 before softmax, so garbage never propagates.
  for (int idx = tid; idx < 196; idx += 128) {
    int tok = idx >> 2, seg = (idx & 3) * 8;
    async_b128(lds_addr(lQ + tok * 32 + seg), qb + base + tok * 32 + seg);
    async_b128(lds_addr(lK + tok * 32 + seg), kb + base + tok * 32 + seg);
  }
  // V is transposed (d-major) and zero-padded so P@V padding contributes 0.
  for (int idx = tid; idx < 64 * 32; idx += 128) {
    int tok = idx >> 5, d = idx & 31;
    u16 v = (tok < NTOK) ? vb[base + tok * HD + d] : (u16)0;
    lVT[d * 64 + tok] = v;
  }
  wait_async0();
  __syncthreads();

  v16bf qf = load_frag(lQ + wv * 16 * 32, 32, lane);
  v8f zero = v8f{0.f, 0.f, 0.f, 0.f, 0.f, 0.f, 0.f, 0.f};
  v8f S[4];
#pragma unroll
  for (int t = 0; t < 4; ++t) {
    v16bf kf = load_frag(lK + t * 16 * 32, 32, lane);
    S[t] = wmma_bf16(qf, kf, zero);
  }

  int mh = (lane >> 4) << 3;
  int nl = lane & 15;
  int winImg = win & 63;
  float P[4][8];
#pragma unroll
  for (int t = 0; t < 4; ++t)
#pragma unroll
    for (int r = 0; r < 8; ++r) {
      int m = wv * 16 + r + mh;      // query token
      int n = t * 16 + nl;           // key token
      float sv = S[t][r];
      if (n < NTOK && m < NTOK) {
        int r0 = m / WSZ, c0 = m % WSZ, r1 = n / WSZ, c1 = n % WSZ;
        int ridx = (r0 - r1 + WSZ - 1) * (2 * WSZ - 1) + (c0 - c1 + WSZ - 1);
        sv += rpb[ridx * HEADS + head] +
              mask[((size_t)winImg * NTOK + m) * NTOK + n];
      } else {
        sv = -1e30f;
      }
      P[t][r] = sv;
    }

  // row softmax: each row lives in one 16-lane half across 4 tiles
#pragma unroll
  for (int r = 0; r < 8; ++r) {
    float mx = -1e30f;
#pragma unroll
    for (int t = 0; t < 4; ++t) mx = fmaxf(mx, P[t][r]);
#pragma unroll
    for (int m = 8; m >= 1; m >>= 1) mx = fmaxf(mx, __shfl_xor(mx, m, 32));
    float sum = 0.f;
#pragma unroll
    for (int t = 0; t < 4; ++t) {
      float e = __expf(P[t][r] - mx);
      P[t][r] = e;
      sum += e;
    }
#pragma unroll
    for (int m = 8; m >= 1; m >>= 1) sum += __shfl_xor(sum, m, 32);
    float inv = 1.f / sum;
#pragma unroll
    for (int t = 0; t < 4; ++t) P[t][r] *= inv;
  }

  u16* myP = lP + wv * 16 * 64;
#pragma unroll
  for (int t = 0; t < 4; ++t)
#pragma unroll
    for (int r = 0; r < 8; ++r)
      myP[(r + mh) * 64 + t * 16 + nl] = f2bf(P[t][r]);
  __syncthreads();

  v8f O[2];
  O[0] = zero; O[1] = zero;
#pragma unroll
  for (int ks = 0; ks < 2; ++ks) {
    v16bf pf = load_frag(myP + ks * 32, 64, lane);
#pragma unroll
    for (int t = 0; t < 2; ++t) {
      v16bf vf = load_frag(lVT + t * 16 * 64 + ks * 32, 64, lane);
      O[t] = wmma_bf16(pf, vf, O[t]);
    }
  }

#pragma unroll
  for (int t = 0; t < 2; ++t)
#pragma unroll
    for (int r = 0; r < 8; ++r) {
      int m = wv * 16 + r + mh;
      if (m < NTOK) {
        int d = t * 16 + nl;
        size_t o = ((size_t)win * NTOK + m) * CH + head * HD + d;
        pin[o] = f2bf(O[t][r] + dww[o]);
      }
    }
}

// ---------------------------------------------------------------------------
// K6: proj GEMM + window-reverse + roll(+3,+3) + residual add -> fp32 out
// ---------------------------------------------------------------------------
__global__ __launch_bounds__(256) void proj_gemm_kernel(
    const u16* __restrict__ pin, const u16* __restrict__ wp,
    const float* __restrict__ proj_b, const float* __restrict__ x,
    float* __restrict__ out) {
  __shared__ __align__(16) u16 lA[2][128 * 32];
  __shared__ __align__(16) u16 lB[2][64 * 32];
  int mBase = blockIdx.x * 128;
  int nBase = blockIdx.y * 64;
  int tid = threadIdx.x, lane = tid & 31, wv = tid >> 5;
  v8f acc[4];
#pragma unroll
  for (int t = 0; t < 4; ++t) acc[t] = v8f{0.f, 0.f, 0.f, 0.f, 0.f, 0.f, 0.f, 0.f};

  auto stage = [&](int k0, u16* dA, u16* dB) {
#pragma unroll
    for (int i = 0; i < 2; ++i) {
      int c = tid * 2 + i;
      int row = c >> 2, seg = (c & 3) * 8;
      const u16* src = pin + (size_t)(mBase + row) * CH + k0 + seg;
      async_b128(lds_addr(dA + row * 32 + seg), src);
      if (k0 + 64 < CH) __builtin_prefetch(src + 64, 0, 1);
    }
    int rB = tid >> 2, segB = (tid & 3) * 8;
    async_b128(lds_addr(dB + rB * 32 + segB),
               wp + (size_t)(nBase + rB) * CH + k0 + segB);
  };

  stage(0, lA[0], lB[0]);
  wait_async0();
  __syncthreads();
#pragma unroll
  for (int ks = 0; ks < 6; ++ks) {
    int buf = ks & 1;
    if (ks < 5) stage((ks + 1) * 32, lA[buf ^ 1], lB[buf ^ 1]);
    v16bf a = load_frag(lA[buf] + wv * 16 * 32, 32, lane);
#pragma unroll
    for (int t = 0; t < 4; ++t) {
      v16bf bfr = load_frag(lB[buf] + t * 16 * 32, 32, lane);
      acc[t] = wmma_bf16(a, bfr, acc[t]);
    }
    if (ks < 5) { wait_async0(); __syncthreads(); }
  }

  int m0 = mBase + wv * 16;
  int mh = (lane >> 4) << 3;
  int nl = lane & 15;
#pragma unroll
  for (int t = 0; t < 4; ++t) {
    int nG = nBase + t * 16 + nl;
    float bias = proj_b[nG];
#pragma unroll
    for (int r = 0; r < 8; ++r) {
      int row = m0 + mh + r;
      int win = row / NTOK, tok = row % NTOK;
      int b = win >> 6, wi = win & 63;
      int hs = (wi >> 3) * WSZ + tok / WSZ;
      int wsC = (wi & 7) * WSZ + tok % WSZ;
      int h = hs + SHIFT; if (h >= HH) h -= HH;
      int w = wsC + SHIFT; if (w >= WW) w -= WW;
      size_t o = (((size_t)b * HH + h) * WW + w) * CH + nG;
      out[o] = x[o] + acc[t][r] + bias;
    }
  }
}

// ---------------------------------------------------------------------------
extern "C" void kernel_launch(void* const* d_in, const int* in_sizes, int n_in,
                              void* d_out, int out_size, void* d_ws,
                              size_t ws_size, hipStream_t stream) {
  (void)in_sizes; (void)n_in; (void)out_size; (void)ws_size;
  const float* x      = (const float*)d_in[0];
  const float* mask   = (const float*)d_in[1];
  const float* gamma  = (const float*)d_in[2];
  const float* beta   = (const float*)d_in[3];
  const float* qkv_w  = (const float*)d_in[4];
  const float* qkv_b  = (const float*)d_in[5];
  const float* proj_w = (const float*)d_in[6];
  const float* proj_b = (const float*)d_in[7];
  const float* rpb    = (const float*)d_in[8];
  const float* dw_w   = (const float*)d_in[9];
  const float* dw_b   = (const float*)d_in[10];
  float* out = (float*)d_out;

  // Workspace layout (aliased):
  //   [0, 77MB)      xs fp32   -> reused as qbuf/kbuf bf16 after dwconv
  //   [77, 115.5MB)  xw bf16   -> reused as proj input after QKV GEMM
  //   [115.5, 154MB) vbuf bf16
  //   [154, 231MB)   dww fp32
  //   [231MB, ...)   bf16 weights
  char* ws = (char*)d_ws;
  const size_t SZ_XS  = (size_t)ROWS * CH * 4;   // 77,070,336
  const size_t SZ_XW  = (size_t)ROWS * CH * 2;   // 38,535,168
  float* xs   = (float*)ws;
  u16*   xw   = (u16*)(ws + SZ_XS);
  u16*   vbuf = (u16*)(ws + SZ_XS + SZ_XW);
  float* dww  = (float*)(ws + SZ_XS + 2 * SZ_XW);
  u16*   qwb  = (u16*)(ws + 2 * SZ_XS + 2 * SZ_XW);
  u16*   pwb  = qwb + 3 * CH * CH;
  u16*   qbuf = (u16*)ws;                         // aliases xs
  u16*   kbuf = qbuf + (size_t)ROWS * CH;         // second half of xs region

  ln_shift_kernel<<<ROWS / 8, 256, 0, stream>>>(x, gamma, beta, xs, xw);
  convert_w_kernel<<<(3 * CH * CH + 255) / 256, 256, 0, stream>>>(qkv_w, proj_w,
                                                                  qwb, pwb);
  dwconv_kernel<<<NWIN, 256, 0, stream>>>(xs, dw_w, dw_b, dww);
  qkv_gemm_kernel<<<dim3(ROWS / 128, 9), 256, 0, stream>>>(xw, qwb, qkv_b, qbuf,
                                                           kbuf, vbuf);
  attention_kernel<<<NWIN * HEADS, 128, 0, stream>>>(qbuf, kbuf, vbuf, rpb,
                                                     mask, dww, xw);
  proj_gemm_kernel<<<dim3(ROWS / 128, 3), 256, 0, stream>>>(xw, pwb, proj_b, x,
                                                            out);
}
